// EGNNConv_21792664060154
// MI455X (gfx1250) — matrix operand order repaired and checked by reading, not compiled, not run on hardware
//
#include <hip/hip_runtime.h>

#define HID  256
#define EDIM 64
#define KP1  608   // 577 padded to 19*32
#define SA1  624   // LDS row stride (bf16 elems) for edge A tile
#define SA2  264   // LDS row stride for 32x256 activation tiles
#define SAN  528   // LDS row stride for node 32x512 A tile
#define MT   32    // rows (edges/nodes) per block: two 16-row M tiles

typedef __bf16 bf16_t;
typedef __attribute__((ext_vector_type(16))) __bf16 v16bf;
typedef __attribute__((ext_vector_type(8)))  __bf16 v8bf;
typedef __attribute__((ext_vector_type(8)))  float  v8f;

__device__ __forceinline__ v8f vzero() {
  v8f z = {0.f, 0.f, 0.f, 0.f, 0.f, 0.f, 0.f, 0.f};
  return z;
}

__device__ __forceinline__ float silu(float v) {
  return v / (1.0f + __expf(-v));
}

__device__ __forceinline__ v8f wmma_bf16(v16bf a, v16bf b, v8f c) {
  // D = A(16x32 bf16) x B(32x16 bf16) + C(16x16 f32)
  return __builtin_amdgcn_wmma_f32_16x16x32_bf16(
      /*neg_a=*/false, a, /*neg_b=*/false, b,
      /*c_mod=*/(short)0, c, /*reuse_a=*/false, /*reuse_b=*/false);
}

// A-matrix fragment in contiguous k order, same as load_b:
// lanes 0-15 (half=0): VGPR0-7 = K k0..k0+15
// lanes 16-31 (half=1): VGPR0-7 = K k0+16..k0+31
__device__ __forceinline__ v16bf load_a(const bf16_t* lds, int mrow, int stride,
                                        int k0, int half) {
  const bf16_t* p = lds + mrow * stride + k0 + half * 16;
  v8bf lo = *(const v8bf*)p;
  v8bf hi = *(const v8bf*)(p + 8);
  return __builtin_shufflevector(lo, hi, 0, 1, 2, 3, 4, 5, 6, 7,
                                 8, 9, 10, 11, 12, 13, 14, 15);
}

// B-matrix fragment in contiguous k order, same as load_a (WT stored [n][kp], row-major):
// lanes 0-15: N=lane, K=k0..k0+15 ; lanes 16-31: N=lane-16, K=k0+16..k0+31
__device__ __forceinline__ v16bf load_b(const bf16_t* __restrict__ wt, int n,
                                        int kp, int k0, int half) {
  return *(const v16bf*)(wt + (long)n * kp + k0 + half * 16);
}

// ---- prep: zero m_i region of d_out, copy x into x_out region ----
__global__ __launch_bounds__(256) void egnn_init_out(
    const float* __restrict__ x, float* __restrict__ out, int N) {
  long i = (long)blockIdx.x * blockDim.x + threadIdx.x;
  long tot = (long)N * HID;
  if (i < tot) out[i] = 0.0f;
  if (i < (long)N * 3) out[tot + i] = x[i];
}

// ---- prep: W (K x Nc, row-major f32) -> WT (Nc x KP, bf16, zero-padded K) ----
__global__ __launch_bounds__(256) void egnn_wt(
    const float* __restrict__ W, bf16_t* __restrict__ WT, int K, int Nc, int KP) {
  long i = (long)blockIdx.x * blockDim.x + threadIdx.x;
  if (i >= (long)Nc * KP) return;
  int n = (int)(i / KP), k = (int)(i % KP);
  WT[i] = (k < K) ? (bf16_t)W[(long)k * Nc + n] : (bf16_t)0.0f;
}

// ---- edge kernel: 32 edges/block, 8 waves, each wave = 2 M-tiles x 2 N-tiles ----
__global__ __launch_bounds__(256) void egnn_edge(
    const float* __restrict__ h, const float* __restrict__ x,
    const int* __restrict__ ei, const float* __restrict__ ea,
    const bf16_t* __restrict__ WT1, const float* __restrict__ be1,
    const bf16_t* __restrict__ WT2, const float* __restrict__ be2,
    const bf16_t* __restrict__ WTc1, const float* __restrict__ bc1,
    const float* __restrict__ Wc2,
    float* __restrict__ mi_out, float* __restrict__ x_out, int E) {
  __shared__ bf16_t Abuf[MT * SA1];
  __shared__ bf16_t A2buf[MT * SA2];
  __shared__ float relp[MT][3];
  __shared__ int dstI[MT];
  __shared__ float coordw[MT];

  const int t = threadIdx.x;
  const int e0 = blockIdx.x * MT;
  const int m = t >> 3;   // 0..31: row within block
  const int c = t & 7;    // 8 threads per row

  int e = e0 + m;
  if (e >= E) e = E - 1;
  const int src = ei[e];
  const int dst = ei[E + e];

  if (c == 0) {
    dstI[m] = dst;
    float rx = x[src * 3 + 0] - x[dst * 3 + 0];
    float ry = x[src * 3 + 1] - x[dst * 3 + 1];
    float rz = x[src * 3 + 2] - x[dst * 3 + 2];
    relp[m][0] = rx; relp[m][1] = ry; relp[m][2] = rz;
    Abuf[m * SA1 + 512] = (bf16_t)(rx * rx + ry * ry + rz * rz);
    coordw[m] = 0.0f;
  } else if (c == 1) {
#pragma unroll
    for (int i = 0; i < 31; ++i)  // zero pad cols 577..607
      Abuf[m * SA1 + 577 + i] = (bf16_t)0.0f;
  }
  {
    const float* hs = h + (long)src * HID;
    const float* hd = h + (long)dst * HID;
#pragma unroll
    for (int i = 0; i < 32; ++i) {
      int col = c * 32 + i;
      Abuf[m * SA1 + col]       = (bf16_t)hs[col];
      Abuf[m * SA1 + 256 + col] = (bf16_t)hd[col];
    }
    const float* eap = ea + (long)e * EDIM;
#pragma unroll
    for (int i = 0; i < 8; ++i)
      Abuf[m * SA1 + 513 + c * 8 + i] = (bf16_t)eap[c * 8 + i];
  }
  __syncthreads();

  const int w = t >> 5, lane = t & 31, half = lane >> 4, ln = lane & 15;
  const int n0a = w * 32, n0b = n0a + 16;

  // GEMM1: (32 x 608) x (608 x 256)
  v8f a00 = vzero(), a01 = vzero(), a10 = vzero(), a11 = vzero();
  for (int kk = 0; kk < KP1; kk += 32) {
    v16bf fa0 = load_a(Abuf, ln, SA1, kk, half);
    v16bf fa1 = load_a(Abuf, 16 + ln, SA1, kk, half);
    v16bf fb0 = load_b(WT1, n0a + ln, KP1, kk, half);
    v16bf fb1 = load_b(WT1, n0b + ln, KP1, kk, half);
    a00 = wmma_bf16(fa0, fb0, a00);
    a01 = wmma_bf16(fa0, fb1, a01);
    a10 = wmma_bf16(fa1, fb0, a10);
    a11 = wmma_bf16(fa1, fb1, a11);
  }
  {
    float bias0 = be1[n0a + ln], bias1 = be1[n0b + ln];
#pragma unroll
    for (int r = 0; r < 8; ++r) {
      int mA = half * 8 + r, mB = 16 + mA;  // C layout: VGPR r -> M=half*8+r, N=ln
      A2buf[mA * SA2 + n0a + ln] = (bf16_t)silu(a00[r] + bias0);
      A2buf[mA * SA2 + n0b + ln] = (bf16_t)silu(a01[r] + bias1);
      A2buf[mB * SA2 + n0a + ln] = (bf16_t)silu(a10[r] + bias0);
      A2buf[mB * SA2 + n0b + ln] = (bf16_t)silu(a11[r] + bias1);
    }
  }
  __syncthreads();

  // GEMM2: (32 x 256) x (256 x 256) -> m_ij
  a00 = vzero(); a01 = vzero(); a10 = vzero(); a11 = vzero();
  for (int kk = 0; kk < HID; kk += 32) {
    v16bf fa0 = load_a(A2buf, ln, SA2, kk, half);
    v16bf fa1 = load_a(A2buf, 16 + ln, SA2, kk, half);
    v16bf fb0 = load_b(WT2, n0a + ln, HID, kk, half);
    v16bf fb1 = load_b(WT2, n0b + ln, HID, kk, half);
    a00 = wmma_bf16(fa0, fb0, a00);
    a01 = wmma_bf16(fa0, fb1, a01);
    a10 = wmma_bf16(fa1, fb0, a10);
    a11 = wmma_bf16(fa1, fb1, a11);
  }
  float f00[8], f01[8], f10[8], f11[8];
  {
    float bias0 = be2[n0a + ln], bias1 = be2[n0b + ln];
#pragma unroll
    for (int r = 0; r < 8; ++r) {
      int mA = half * 8 + r, mB = 16 + mA;
      f00[r] = silu(a00[r] + bias0);
      f01[r] = silu(a01[r] + bias1);
      f10[r] = silu(a10[r] + bias0);
      f11[r] = silu(a11[r] + bias1);
      if (e0 + mA < E) {  // segment_sum(m_ij, dst) -> m_i (L2-resident atomics)
        long dr = (long)dstI[mA] * HID;
        unsafeAtomicAdd(&mi_out[dr + n0a + ln], f00[r]);
        unsafeAtomicAdd(&mi_out[dr + n0b + ln], f01[r]);
      }
      if (e0 + mB < E) {
        long dr = (long)dstI[mB] * HID;
        unsafeAtomicAdd(&mi_out[dr + n0a + ln], f10[r]);
        unsafeAtomicAdd(&mi_out[dr + n0b + ln], f11[r]);
      }
    }
  }
  __syncthreads();
#pragma unroll
  for (int r = 0; r < 8; ++r) {
    int mA = half * 8 + r, mB = 16 + mA;
    A2buf[mA * SA2 + n0a + ln] = (bf16_t)f00[r];
    A2buf[mA * SA2 + n0b + ln] = (bf16_t)f01[r];
    A2buf[mB * SA2 + n0a + ln] = (bf16_t)f10[r];
    A2buf[mB * SA2 + n0b + ln] = (bf16_t)f11[r];
  }
  __syncthreads();

  // GEMM3: coord MLP layer 1, then dot with Wc2 and lane-reduce
  a00 = vzero(); a01 = vzero(); a10 = vzero(); a11 = vzero();
  for (int kk = 0; kk < HID; kk += 32) {
    v16bf fa0 = load_a(A2buf, ln, SA2, kk, half);
    v16bf fa1 = load_a(A2buf, 16 + ln, SA2, kk, half);
    v16bf fb0 = load_b(WTc1, n0a + ln, HID, kk, half);
    v16bf fb1 = load_b(WTc1, n0b + ln, HID, kk, half);
    a00 = wmma_bf16(fa0, fb0, a00);
    a01 = wmma_bf16(fa0, fb1, a01);
    a10 = wmma_bf16(fa1, fb0, a10);
    a11 = wmma_bf16(fa1, fb1, a11);
  }
  {
    float bias0 = bc1[n0a + ln], bias1 = bc1[n0b + ln];
    float wc0 = Wc2[n0a + ln],  wc1 = Wc2[n0b + ln];
#pragma unroll
    for (int r = 0; r < 8; ++r) {
      float pA = silu(a00[r] + bias0) * wc0 + silu(a01[r] + bias1) * wc1;
      float pB = silu(a10[r] + bias0) * wc0 + silu(a11[r] + bias1) * wc1;
#pragma unroll
      for (int off = 1; off < 16; off <<= 1) {
        pA += __shfl_xor(pA, off, 32);
        pB += __shfl_xor(pB, off, 32);
      }
      if (ln == 0) {
        atomicAdd(&coordw[half * 8 + r], pA);
        atomicAdd(&coordw[16 + half * 8 + r], pB);
      }
    }
  }
  __syncthreads();
  if (t < MT) {
    int mm = t;
    if (e0 + mm < E) {
      float cw = coordw[mm];
      long db = (long)dstI[mm] * 3;
      unsafeAtomicAdd(&x_out[db + 0], relp[mm][0] * cw);
      unsafeAtomicAdd(&x_out[db + 1], relp[mm][1] * cw);
      unsafeAtomicAdd(&x_out[db + 2], relp[mm][2] * cw);
    }
  }
}

// ---- node kernel: h_out = h + silu([h, m_i] Wn1 + bn1) Wn2 + bn2 ----
// hio holds m_i on entry; each block reads its own rows, then overwrites them.
__global__ __launch_bounds__(256) void egnn_node(
    const float* __restrict__ h,
    const bf16_t* __restrict__ WTn1, const float* __restrict__ bn1,
    const bf16_t* __restrict__ WTn2, const float* __restrict__ bn2,
    float* __restrict__ hio, int N) {
  __shared__ bf16_t Abuf[MT * SAN];
  __shared__ bf16_t A2buf[MT * SA2];
  const int t = threadIdx.x;
  const int r0 = blockIdx.x * MT;
  const int m = t >> 3, c = t & 7;
  int row = r0 + m;
  if (row >= N) row = N - 1;
  {
    const float* hr = h + (long)row * HID;
    const float* mr = hio + (long)row * HID;
#pragma unroll
    for (int i = 0; i < 32; ++i) {
      int col = c * 32 + i;
      Abuf[m * SAN + col]       = (bf16_t)hr[col];
      Abuf[m * SAN + 256 + col] = (bf16_t)mr[col];
    }
  }
  __syncthreads();

  const int w = t >> 5, lane = t & 31, half = lane >> 4, ln = lane & 15;
  const int n0a = w * 32, n0b = n0a + 16;

  v8f a00 = vzero(), a01 = vzero(), a10 = vzero(), a11 = vzero();
  for (int kk = 0; kk < 2 * HID; kk += 32) {
    v16bf fa0 = load_a(Abuf, ln, SAN, kk, half);
    v16bf fa1 = load_a(Abuf, 16 + ln, SAN, kk, half);
    v16bf fb0 = load_b(WTn1, n0a + ln, 2 * HID, kk, half);
    v16bf fb1 = load_b(WTn1, n0b + ln, 2 * HID, kk, half);
    a00 = wmma_bf16(fa0, fb0, a00);
    a01 = wmma_bf16(fa0, fb1, a01);
    a10 = wmma_bf16(fa1, fb0, a10);
    a11 = wmma_bf16(fa1, fb1, a11);
  }
  {
    float bias0 = bn1[n0a + ln], bias1 = bn1[n0b + ln];
#pragma unroll
    for (int r = 0; r < 8; ++r) {
      int mA = half * 8 + r, mB = 16 + mA;
      A2buf[mA * SA2 + n0a + ln] = (bf16_t)silu(a00[r] + bias0);
      A2buf[mA * SA2 + n0b + ln] = (bf16_t)silu(a01[r] + bias1);
      A2buf[mB * SA2 + n0a + ln] = (bf16_t)silu(a10[r] + bias0);
      A2buf[mB * SA2 + n0b + ln] = (bf16_t)silu(a11[r] + bias1);
    }
  }
  __syncthreads();

  a00 = vzero(); a01 = vzero(); a10 = vzero(); a11 = vzero();
  for (int kk = 0; kk < HID; kk += 32) {
    v16bf fa0 = load_a(A2buf, ln, SA2, kk, half);
    v16bf fa1 = load_a(A2buf, 16 + ln, SA2, kk, half);
    v16bf fb0 = load_b(WTn2, n0a + ln, HID, kk, half);
    v16bf fb1 = load_b(WTn2, n0b + ln, HID, kk, half);
    a00 = wmma_bf16(fa0, fb0, a00);
    a01 = wmma_bf16(fa0, fb1, a01);
    a10 = wmma_bf16(fa1, fb0, a10);
    a11 = wmma_bf16(fa1, fb1, a11);
  }
  {
    float bias0 = bn2[n0a + ln], bias1 = bn2[n0b + ln];
#pragma unroll
    for (int r = 0; r < 8; ++r) {
      int mA = half * 8 + r, mB = 16 + mA;
      int gA = r0 + mA, gB = r0 + mB;
      if (gA < N) {
        long gb = (long)gA * HID;
        hio[gb + n0a + ln] = a00[r] + bias0 + h[gb + n0a + ln];
        hio[gb + n0b + ln] = a01[r] + bias1 + h[gb + n0b + ln];
      }
      if (gB < N) {
        long gb = (long)gB * HID;
        hio[gb + n0a + ln] = a10[r] + bias0 + h[gb + n0a + ln];
        hio[gb + n0b + ln] = a11[r] + bias1 + h[gb + n0b + ln];
      }
    }
  }
}

extern "C" void kernel_launch(void* const* d_in, const int* in_sizes, int n_in,
                              void* d_out, int out_size, void* d_ws, size_t ws_size,
                              hipStream_t stream) {
  const float* h   = (const float*)d_in[0];
  const float* x   = (const float*)d_in[1];
  const int*   ei  = (const int*)d_in[2];
  const float* ea  = (const float*)d_in[3];
  const float* We1 = (const float*)d_in[4];
  const float* be1 = (const float*)d_in[5];
  const float* We2 = (const float*)d_in[6];
  const float* be2 = (const float*)d_in[7];
  const float* Wn1 = (const float*)d_in[8];
  const float* bn1 = (const float*)d_in[9];
  const float* Wn2 = (const float*)d_in[10];
  const float* bn2 = (const float*)d_in[11];
  const float* Wc1 = (const float*)d_in[12];
  const float* bc1 = (const float*)d_in[13];
  const float* Wc2 = (const float*)d_in[14];

  const int N = in_sizes[0] / HID;
  const int E = in_sizes[2] / 2;

  float* out = (float*)d_out;
  float* mi  = out;                   // N*HID: m_i accumulator, becomes h_out
  float* xo  = out + (long)N * HID;   // N*3:   x_out accumulator

  // bf16 transposed weights in workspace (all 32B-aligned offsets)
  char* ws = (char*)d_ws;
  bf16_t* WT1  = (bf16_t*)(ws + 0);        // 256 x 608 -> 311296 B
  bf16_t* WT2  = (bf16_t*)(ws + 311296);   // 256 x 256 -> 131072 B
  bf16_t* WTc1 = (bf16_t*)(ws + 442368);   // 256 x 256 -> 131072 B
  bf16_t* WTn1 = (bf16_t*)(ws + 573440);   // 256 x 512 -> 262144 B
  bf16_t* WTn2 = (bf16_t*)(ws + 835584);   // 256 x 256 -> 131072 B

  long tot = (long)N * HID;
  egnn_init_out<<<(int)((tot + 255) / 256), 256, 0, stream>>>(x, out, N);

  egnn_wt<<<(256 * KP1 + 255) / 256, 256, 0, stream>>>(We1, WT1, 577, HID, KP1);
  egnn_wt<<<(256 * 256 + 255) / 256, 256, 0, stream>>>(We2, WT2, HID, HID, HID);
  egnn_wt<<<(256 * 256 + 255) / 256, 256, 0, stream>>>(Wc1, WTc1, HID, HID, HID);
  egnn_wt<<<(256 * 512 + 255) / 256, 256, 0, stream>>>(Wn1, WTn1, 2 * HID, HID, 2 * HID);
  egnn_wt<<<(256 * 256 + 255) / 256, 256, 0, stream>>>(Wn2, WTn2, HID, HID, HID);

  egnn_edge<<<(E + MT - 1) / MT, 256, 0, stream>>>(h, x, ei, ea, WT1, be1, WT2, be2,
                                                   WTc1, bc1, Wc2, mi, xo, E);
  egnn_node<<<(N + MT - 1) / MT, 256, 0, stream>>>(h, WTn1, bn1, WTn2, bn2, mi, N);
}
